// MessagePassing_24936580120614
// MI455X (gfx1250) — matrix operand order repaired and checked by reading, not compiled, not run on hardware
//
#include <hip/hip_runtime.h>
#include <hip/hip_bf16.h>
#include <math.h>

typedef __attribute__((ext_vector_type(4)))  float        v4f;
typedef __attribute__((ext_vector_type(8)))  float        v8f;
typedef __attribute__((ext_vector_type(16))) __bf16       v16bf;
typedef __attribute__((ext_vector_type(8)))  unsigned int v8u;

#define H_DIM 128
#define F_DIM 128
#define TH3   384
#define INV3  0.57735026918962576f   // 1/sqrt(3)
#define INVH  0.08838834764831845f   // 1/sqrt(128)
#define SSCALE (1.0f/0.6f)

// ---- fp32 -> bf16 conversion helpers (RNE) ----
__device__ __forceinline__ unsigned int f2bf2(float a, float b) {
#if __has_builtin(__builtin_amdgcn_cvt_pk_bf16_f32)
  typedef __attribute__((ext_vector_type(2))) __bf16 v2bf;
  v2bf p = __builtin_amdgcn_cvt_pk_bf16_f32(a, b);
  return __builtin_bit_cast(unsigned int, p);
#else
  unsigned int ua = __builtin_bit_cast(unsigned int, a);
  unsigned int ub = __builtin_bit_cast(unsigned int, b);
  ua = (ua + 0x7FFFu + ((ua >> 16) & 1u)) >> 16;
  ub = (ub + 0x7FFFu + ((ub >> 16) & 1u)) & 0xFFFF0000u;
  return ua | ub;
#endif
}

__device__ __forceinline__ __bf16 f2bf(float f) {
  unsigned int u = __builtin_bit_cast(unsigned int, f);
  unsigned int r = (u + 0x7FFFu + ((u >> 16) & 1u)) >> 16;
  unsigned short hs = (unsigned short)r;
  return __builtin_bit_cast(__bf16, hs);
}

__device__ __forceinline__ float fast_rcp(float x) {
#if __has_builtin(__builtin_amdgcn_rcpf)
  return __builtin_amdgcn_rcpf(x);   // v_rcp_f32
#else
  return 1.0f / x;
#endif
}

__device__ __forceinline__ v8f wmma_bf16(v16bf a, v16bf b, v8f c) {
  return __builtin_amdgcn_wmma_f32_16x16x32_bf16(false, a, false, b, (short)0, c, false, false);
}

// Pack 16 A-layout elements (elems 0..7 <- f[0..7], 8..15 <- f[16..23]) given
// four float4s loaded from the two 32B runs.
__device__ __forceinline__ v8u pack_a_vals(v4f a0, v4f a1, v4f b0, v4f b1) {
  v8u u;
  u[0] = f2bf2(a0[0], a0[1]); u[1] = f2bf2(a0[2], a0[3]);
  u[2] = f2bf2(a1[0], a1[1]); u[3] = f2bf2(a1[2], a1[3]);
  u[4] = f2bf2(b0[0], b0[1]); u[5] = f2bf2(b0[2], b0[3]);
  u[6] = f2bf2(b1[0], b1[1]); u[7] = f2bf2(b1[2], b1[3]);
  return u;
}

// Streaming (non-temporal) variant for single-use inputs: keep the 192MB L2 for
// the gather/scatter working set (x_h, vec, outputs).
__device__ __forceinline__ v8u pack_a_row_nt(const float* rp) {
  v4f a0 = __builtin_nontemporal_load((const v4f*)(rp + 0));
  v4f a1 = __builtin_nontemporal_load((const v4f*)(rp + 4));
  v4f b0 = __builtin_nontemporal_load((const v4f*)(rp + 16));
  v4f b1 = __builtin_nontemporal_load((const v4f*)(rp + 20));
  return pack_a_vals(a0, a1, b0, b1);
}

__device__ __forceinline__ v8u pack_a_row(const float* rp) {
  v8u u;
#pragma unroll
  for (int t = 0; t < 4; ++t) u[t] = f2bf2(rp[2 * t], rp[2 * t + 1]);
#pragma unroll
  for (int t = 0; t < 4; ++t) u[4 + t] = f2bf2(rp[16 + 2 * t], rp[17 + 2 * t]);
  return u;
}

__global__ void zero_kernel(float* p, long long n) {
  long long i = (long long)blockIdx.x * blockDim.x + threadIdx.x;
  if (i < n) p[i] = 0.0f;
}

// Pack fp32 KxN matrix into bf16 WMMA B-fragments: frag = n*(K/32)+kk;
// within frag element t: lane = t/16, k = t%16; row = kk*32 + (lane>=16?8:0) + (k&7) + (k>=8?16:0), col = n*16 + lane%16.
__global__ void pack_b(const float* __restrict__ src, unsigned short* __restrict__ dstu,
                       int K, int Ncols) {
  __bf16* dst = (__bf16*)dstu;
  int t = blockIdx.x * blockDim.x + threadIdx.x;
  if (t >= K * Ncols) return;
  int frag = t >> 9, wf = t & 511;
  int lane = wf >> 4, k = wf & 15;
  int kchunks = K >> 5;
  int n = frag / kchunks, kk = frag - n * kchunks;
  int half = lane >> 4, lm = lane & 15;
  int row = kk * 32 + half * 8 + (k & 7) + (k >> 3) * 16;
  int col = n * 16 + lm;
  dst[t] = f2bf(src[row * Ncols + col]);
}

// x_h = (ScaledSiLU(x@W1+b1)) @ W2 + b2 ; one 16-node tile per block, 4 waves.
__global__ void node_kernel(const float* __restrict__ x,
                            const unsigned short* __restrict__ w1u, const float* __restrict__ b1,
                            const unsigned short* __restrict__ w2u, const float* __restrict__ b2,
                            float* __restrict__ xh) {
  const __bf16* w1bf = (const __bf16*)w1u;
  const __bf16* w2bf = (const __bf16*)w2u;
  __shared__ __align__(32) unsigned int aLds[4 * 32 * 8];  // [kk][lane][dword]
  __shared__ float hl[16 * 64];
  int lane = threadIdx.x & 31, w = threadIdx.x >> 5;
  int half = lane >> 4, lm = lane & 15;
  int nbase = blockIdx.x * 16;

  // Cooperative A staging for GEMM1: wave w builds K-chunk kk=w (x is single-use: NT)
  {
    const float* rp = x + (size_t)(nbase + lm) * H_DIM + w * 32 + half * 8;
    *reinterpret_cast<v8u*>(&aLds[(w * 32 + lane) * 8]) = pack_a_row_nt(rp);
  }
  __syncthreads();

  // GEMM1: [16,128] @ [128,64]; wave w owns N-tile w
  v8f acc1 = {0.f, 0.f, 0.f, 0.f, 0.f, 0.f, 0.f, 0.f};
#pragma unroll
  for (int kk = 0; kk < 4; ++kk) {
    v16bf a = *reinterpret_cast<const v16bf*>(&aLds[(kk * 32 + lane) * 8]);
    v16bf b = *reinterpret_cast<const v16bf*>(w1bf + (w * 4 + kk) * 512 + lane * 16);
    acc1 = wmma_bf16(a, b, acc1);
  }
#pragma unroll
  for (int rr = 0; rr < 8; ++rr) {
    int row = rr + (half ? 8 : 0);
    int col = 16 * w + lm;
    float v = acc1[rr] + b1[col];
    // ScaledSiLU: v_exp_f32 + v_rcp_f32 (no IEEE divide expansion)
    hl[row * 64 + col] = v * SSCALE * fast_rcp(1.0f + __expf(-v));
  }
  __syncthreads();

  // GEMM2: [16,64] @ [64,384]; wave w owns N-tiles 6w..6w+5
  v8f acc2[6];
#pragma unroll
  for (int t6 = 0; t6 < 6; ++t6) acc2[t6] = (v8f){0.f, 0.f, 0.f, 0.f, 0.f, 0.f, 0.f, 0.f};
#pragma unroll
  for (int kk = 0; kk < 2; ++kk) {
    v8u u = pack_a_row(hl + lm * 64 + kk * 32 + half * 8);
    v16bf a = __builtin_bit_cast(v16bf, u);
#pragma unroll
    for (int t6 = 0; t6 < 6; ++t6) {
      int n = 6 * w + t6;
      v16bf b = *reinterpret_cast<const v16bf*>(w2bf + (n * 2 + kk) * 512 + lane * 16);
      acc2[t6] = wmma_bf16(a, b, acc2[t6]);
    }
  }
#pragma unroll
  for (int t6 = 0; t6 < 6; ++t6) {
    int n = 6 * w + t6;
    int col = 16 * n + lm;
#pragma unroll
    for (int rr = 0; rr < 8; ++rr) {
      int row = rr + (half ? 8 : 0);
      xh[(size_t)(nbase + row) * TH3 + col] = acc2[t6][rr] + b2[col];
    }
  }
}

// Fused: rbf_h = edge_rbf@We + be ; xm = x_h[j]*rbf_h/sqrt3 ; vec_ji ; atomic scatter-add.
// One 16-edge tile per block, 4 waves; wave w owns N-tiles {w+4r+8c : r<2, c<3}
// so chunks c=0,1,2 (x_ji1/2/3) share lane/element coordinates (same h-column).
__global__ void edge_kernel(const float* __restrict__ edge_rbf,
                            const int* __restrict__ ei,
                            const float* __restrict__ edge_vec,
                            const float* __restrict__ xh,
                            const unsigned short* __restrict__ webf_u,
                            const float* __restrict__ be,
                            const float* __restrict__ vec,
                            float* __restrict__ dx, float* __restrict__ dv, int E) {
  const __bf16* webf = reinterpret_cast<const __bf16*>(webf_u);
  __shared__ __align__(32) unsigned int aLds[4 * 32 * 8];  // [kk][lane][dword]
  int lane = threadIdx.x & 31, w = threadIdx.x >> 5;
  int half = lane >> 4, lm = lane & 15;
  int ebase = blockIdx.x * 16;

  // Cooperative A staging: wave w builds K-chunk kk=w (edge_rbf is single-use: NT)
  {
    const float* rp = edge_rbf + (size_t)(ebase + lm) * F_DIM + w * 32 + half * 8;
    *reinterpret_cast<v8u*>(&aLds[(w * 32 + lane) * 8]) = pack_a_row_nt(rp);
  }
  __syncthreads();

  v8f acc[3][2];
#pragma unroll
  for (int c = 0; c < 3; ++c)
#pragma unroll
    for (int r = 0; r < 2; ++r) acc[c][r] = (v8f){0.f, 0.f, 0.f, 0.f, 0.f, 0.f, 0.f, 0.f};

#pragma unroll
  for (int kk = 0; kk < 4; ++kk) {
    v16bf a = *reinterpret_cast<const v16bf*>(&aLds[(kk * 32 + lane) * 8]);
#pragma unroll
    for (int c = 0; c < 3; ++c)
#pragma unroll
      for (int r = 0; r < 2; ++r) {
        int n = w + 4 * r + 8 * c;
        v16bf b = *reinterpret_cast<const v16bf*>(webf + (n * 4 + kk) * 512 + lane * 16);
        acc[c][r] = wmma_bf16(a, b, acc[c][r]);
      }
  }

  float beL[3][2];
#pragma unroll
  for (int r = 0; r < 2; ++r) {
    int hc = 16 * w + 64 * r + lm;
    beL[0][r] = be[hc];
    beL[1][r] = be[128 + hc];
    beL[2][r] = be[256 + hc];
  }

#pragma unroll
  for (int rr = 0; rr < 8; ++rr) {
    int e = ebase + rr + (half ? 8 : 0);
    int jn = ei[e];
    int ii = ei[E + e];
    float ev0 = edge_vec[3 * e + 0], ev1 = edge_vec[3 * e + 1], ev2 = edge_vec[3 * e + 2];
    const float* xhj = xh + (size_t)jn * TH3;
    const float* vj  = vec + (size_t)jn * TH3;
    float* dvo = dv + (size_t)ii * TH3;
    float* dxo = dx + (size_t)ii * H_DIM;
#pragma unroll
    for (int r = 0; r < 2; ++r) {
      int hc = 16 * w + 64 * r + lm;
      float xm1 = (acc[0][r][rr] + beL[0][r]) * xhj[hc]       * INV3;
      float xm2 = (acc[1][r][rr] + beL[1][r]) * xhj[128 + hc] * INV3;
      float xm3 = (acc[2][r][rr] + beL[2][r]) * xhj[256 + hc] * INV3;
      atomicAdd(dxo + hc, xm3);
      atomicAdd(dvo + hc,       (xm1 * vj[hc]       + xm2 * ev0) * INVH);
      atomicAdd(dvo + 128 + hc, (xm1 * vj[128 + hc] + xm2 * ev1) * INVH);
      atomicAdd(dvo + 256 + hc, (xm1 * vj[256 + hc] + xm2 * ev2) * INVH);
    }
  }
}

extern "C" void kernel_launch(void* const* d_in, const int* in_sizes, int n_in,
                              void* d_out, int out_size, void* d_ws, size_t ws_size,
                              hipStream_t stream) {
  (void)n_in; (void)ws_size;
  const float* x    = (const float*)d_in[0];
  const float* vec  = (const float*)d_in[1];
  const float* erbf = (const float*)d_in[2];
  const float* evec = (const float*)d_in[3];
  const int*   ei   = (const int*)d_in[4];
  const float* W1   = (const float*)d_in[5];
  const float* b1   = (const float*)d_in[6];
  const float* W2   = (const float*)d_in[7];
  const float* b2   = (const float*)d_in[8];
  const float* We   = (const float*)d_in[9];
  const float* be   = (const float*)d_in[10];

  int N = in_sizes[0] / H_DIM;
  int E = in_sizes[3] / 3;

  // workspace layout (all 32B aligned)
  float* xh = (float*)d_ws;
  size_t off = (size_t)N * TH3 * sizeof(float);
  unsigned short* w1bf = (unsigned short*)((char*)d_ws + off); off += (size_t)H_DIM * (H_DIM / 2) * 2;
  unsigned short* w2bf = (unsigned short*)((char*)d_ws + off); off += (size_t)(H_DIM / 2) * TH3 * 2;
  unsigned short* webf = (unsigned short*)((char*)d_ws + off);

  long long nz = (long long)out_size;
  zero_kernel<<<(unsigned)((nz + 255) / 256), 256, 0, stream>>>((float*)d_out, nz);

  pack_b<<<(H_DIM * (H_DIM / 2) + 255) / 256, 256, 0, stream>>>(W1, w1bf, H_DIM, H_DIM / 2);
  pack_b<<<((H_DIM / 2) * TH3 + 255) / 256, 256, 0, stream>>>(W2, w2bf, H_DIM / 2, TH3);
  pack_b<<<(F_DIM * TH3 + 255) / 256, 256, 0, stream>>>(We, webf, F_DIM, TH3);

  node_kernel<<<N / 16, 128, 0, stream>>>(x, w1bf, b1, w2bf, b2, xh);

  float* dx = (float*)d_out;
  float* dv = dx + (size_t)N * H_DIM;
  edge_kernel<<<E / 16, 128, 0, stream>>>(erbf, ei, evec, xh, webf, be, vec, dx, dv, E);
}
